// MaskedDeepRNN_6940667150753
// MI455X (gfx1250) — compile-verified, matrix operand order checked by reading, not run on hardware
//
#include <hip/hip_runtime.h>

// ---------------------------------------------------------------------------
// MaskedDeepRNN (2-layer LSTM) for MI455X / gfx1250, wave32 + WMMA bf16.
// B=64, T=512, D_IN=256, H=512, G=4H=2048.
//
// v2: weights-LDS-resident recurrence.
//  * 64 workgroups = 4 batch tiles x 16 hidden-column groups.
//    WG (bt,g) owns hidden cols [g*32, g*32+32) == k-chunk g of next step's
//    A-fragments, plus the 4 matching i/f/g/o gate slices -> cell update is
//    local. Its W_hh / W_ih fragment slices (128KB + 64..128KB bf16) live in
//    LDS for all 512 steps (CDNA5: 320KB LDS per WGP).
//  * h exchanged through a double-buffered global buffer (step parity);
//    one device-scope generation barrier per step among the 16 WGs of a
//    batch group (cluster-shaped topology; implemented with agent-scope
//    atomics + s_sleep spin since HIP has no verified cluster dispatch).
//  * Input projection fused into the K-loop (K = 512 + Din per step).
// ---------------------------------------------------------------------------

typedef __attribute__((ext_vector_type(16))) __bf16 v16bf;
typedef __attribute__((ext_vector_type(8)))  float  v8f;

#define BATCH 64
#define TSTEPS 512
#define DIN 256
#define HID 512
#define GDIM 2048

__device__ __forceinline__ v8f wmma_bf16(v16bf a, v16bf b, v8f c) {
  return __builtin_amdgcn_wmma_f32_16x16x32_bf16(
      /*neg_a=*/false, a, /*neg_b=*/false, b,
      /*c_mod=*/(short)0, c, /*reuse_a=*/false, /*reuse_b=*/false);
}

// Fragment element mapping (ISA 7.12.2, 16-bit operands, wave32):
//   lane in [0,16):  row = lane,     k-offsets {0..7} U {16..23}
//   lane in [16,32): row = lane-16,  k-offsets {8..15} U {24..31}
//   element e: half = e>>3, kk = e&7
__device__ __forceinline__ int frag_k(int kc, int lane, int e) {
  int half = e >> 3;
  int kk   = e & 7;
  return kc * 32 + half * 16 + ((lane < 16) ? 0 : 8) + kk;
}

// ---------------------------------------------------------------------------
// Pack masked weight W[N x K] into B-fragment blobs out[nt][kc][lane][16].
// ---------------------------------------------------------------------------
__global__ void pack_weight_bf16(const float* __restrict__ W,
                                 const float* __restrict__ Msk,
                                 __bf16* __restrict__ out, int K, int N) {
  int idx = blockIdx.x * blockDim.x + threadIdx.x;
  if (idx >= N * K) return;
  int e    = idx & 15;
  int lane = (idx >> 4) & 31;
  int f    = idx >> 9;
  int numKc = K >> 5;
  int kc = f % numKc;
  int nt = f / numKc;
  int k  = frag_k(kc, lane, e);
  int n  = nt * 16 + (lane & 15);
  out[idx] = (__bf16)(W[n * K + k] * Msk[n * K + k]);
}

__global__ void pack_bias(const float* __restrict__ bi,
                          const float* __restrict__ bh,
                          float* __restrict__ out, int n) {
  int i = blockIdx.x * blockDim.x + threadIdx.x;
  if (i < n) out[i] = bi[i] + bh[i];
}

// Pack x[B][T][DIN] fp32 into A-fragment blobs out[btile][t][kc][lane][16].
__global__ void pack_x_bf16(const float* __restrict__ x,
                            __bf16* __restrict__ out) {
  int idx = blockIdx.x * blockDim.x + threadIdx.x;
  if (idx >= BATCH * TSTEPS * DIN) return;
  int e    = idx & 15;
  int lane = (idx >> 4) & 31;
  int f    = idx >> 9;
  const int numKc = DIN >> 5;   // 8
  int kc    = f % numKc;
  int rem   = f / numKc;
  int t     = rem % TSTEPS;
  int btile = rem / TSTEPS;
  int k = frag_k(kc, lane, e);
  int b = btile * 16 + (lane & 15);
  out[idx] = (__bf16)x[((size_t)b * TSTEPS + t) * DIN + k];
}

__global__ void zero_words(unsigned* __restrict__ p, int n) {
  int i = blockIdx.x * blockDim.x + threadIdx.x;
  if (i < n) p[i] = 0u;
}

// ---------------------------------------------------------------------------
// Generation barrier among NWG workgroups (bar[0]=arrivals, bar[1]=generation)
// ---------------------------------------------------------------------------
__device__ __forceinline__ void group_barrier(unsigned* bar, unsigned nwg) {
  __syncthreads();
  __threadfence();   // release: h stores visible device-wide
  if (threadIdx.x == 0) {
    unsigned gen = __hip_atomic_load(&bar[1], __ATOMIC_RELAXED,
                                     __HIP_MEMORY_SCOPE_AGENT);
    unsigned a = __hip_atomic_fetch_add(&bar[0], 1u, __ATOMIC_ACQ_REL,
                                        __HIP_MEMORY_SCOPE_AGENT);
    if (a == nwg - 1u) {
      __hip_atomic_store(&bar[0], 0u, __ATOMIC_RELAXED,
                         __HIP_MEMORY_SCOPE_AGENT);
      __hip_atomic_fetch_add(&bar[1], 1u, __ATOMIC_RELEASE,
                             __HIP_MEMORY_SCOPE_AGENT);
    } else {
      while (__hip_atomic_load(&bar[1], __ATOMIC_ACQUIRE,
                               __HIP_MEMORY_SCOPE_AGENT) == gen)
        __builtin_amdgcn_s_sleep(1);
    }
  }
  __syncthreads();
  __threadfence();   // acquire: drop stale h lines before next step's reads
}

// ---------------------------------------------------------------------------
// Weights-resident LSTM layer. Grid = 64 WGs (bt = blk&3, g = blk>>2),
// 64 threads (2 waves). Wave w owns hidden cols [g*32+w*16, +16).
//   hspin: [bt][parity][16 kc][512] bf16 A-fragment double buffer
//   xp   : [bt][t][KCX][512] A-fragments (x for L0, h0 for L1)
//   hout : [bt][t][16][512]  (layer 0 output, A-fragments)
// ---------------------------------------------------------------------------
template <int KCX, bool FINAL>
__global__ __launch_bounds__(64, 1)
void lstm_recur(const __bf16* __restrict__ Whh,   // [128][16][512] B-frags
                const __bf16* __restrict__ Wih,   // [128][KCX][512] B-frags
                const __bf16* __restrict__ xp,
                const float*  __restrict__ bsum,  // [4H]
                __bf16* __restrict__ hspin,
                unsigned* __restrict__ bars,
                __bf16* __restrict__ hout,
                float*  __restrict__ dout)
{
  __shared__ __align__(16) __bf16 whh_l[2 * 4 * 16 * 512];   // 128 KB
  __shared__ __align__(16) __bf16 wih_l[2 * 4 * KCX * 512];  // 64/128 KB

  const int bt   = blockIdx.x & 3;
  const int g    = blockIdx.x >> 2;    // hidden-column group 0..15
  const int tid  = threadIdx.x;
  const int w    = tid >> 5;           // wave 0..1
  const int lane = tid & 31;
  unsigned* bar  = bars + bt * 64;     // one 256B line per batch group

  // ---- stage this WG's weight slices into LDS (once, 16B units) ----
  for (int u = tid; u < (2 * 4 * 16 * 512) / 8; u += 64) {
    int frag = u >> 6, within = u & 63;
    int kc = frag & 15, q = (frag >> 4) & 3, jt = frag >> 6;
    int nt = q * 32 + g * 2 + jt;
    ((uint4*)whh_l)[u] =
        ((const uint4*)(Whh + ((size_t)(nt * 16 + kc) << 9)))[within];
  }
  for (int u = tid; u < (2 * 4 * KCX * 512) / 8; u += 64) {
    int frag = u >> 6, within = u & 63;
    int kc = frag % KCX, tmp = frag / KCX;
    int q = tmp & 3, jt = tmp >> 2;
    int nt = q * 32 + g * 2 + jt;
    ((uint4*)wih_l)[u] =
        ((const uint4*)(Wih + ((size_t)(nt * KCX + kc) << 9)))[within];
  }
  __syncthreads();

  const __bf16* wh = whh_l + (size_t)w * 4 * 16 * 512;
  const __bf16* wi = wih_l + (size_t)w * 4 * KCX * 512;

  const int lm = lane & 15;
  const int j  = g * 32 + w * 16 + lm;     // hidden column owned by this lane
  const float bI = bsum[0 * HID + j];
  const float bF = bsum[1 * HID + j];
  const float bG = bsum[2 * HID + j];
  const float bO = bsum[3 * HID + j];

  // h write-back scatter (A-fragment mapping); this WG writes k-chunk g.
  const int r5 = w * 16 + lm;
  const int ew = (r5 & 7) + ((r5 >= 16) ? 8 : 0);
  const int lp = ((r5 >> 3) & 1) ? 16 : 0;

  float cc[8];
#pragma unroll
  for (int r = 0; r < 8; ++r) cc[r] = 0.f;

  const __bf16* xbt = xp + (size_t)bt * TSTEPS * KCX * 512;

  for (int t = 0; t < TSTEPS; ++t) {
    const __bf16* hr = hspin + ((size_t)(bt * 2 + (t & 1)) * 16) * 512;
    v8f a0 = {}, a1 = {}, a2 = {}, a3 = {};

    // --- recurrent projection: full h (16 chunks) x LDS-resident Whh ---
#pragma unroll 4
    for (int kc = 0; kc < 16; ++kc) {
      v16bf a  = *(const v16bf*)(hr + kc * 512 + lane * 16);
      v16bf b0 = *(const v16bf*)(wh + (0 * 16 + kc) * 512 + lane * 16);
      v16bf b1 = *(const v16bf*)(wh + (1 * 16 + kc) * 512 + lane * 16);
      v16bf b2 = *(const v16bf*)(wh + (2 * 16 + kc) * 512 + lane * 16);
      v16bf b3 = *(const v16bf*)(wh + (3 * 16 + kc) * 512 + lane * 16);
      a0 = wmma_bf16(a, b0, a0);
      a1 = wmma_bf16(a, b1, a1);
      a2 = wmma_bf16(a, b2, a2);
      a3 = wmma_bf16(a, b3, a3);
    }
    // --- fused input projection ---
    const __bf16* xt = xbt + (size_t)t * KCX * 512;
#pragma unroll 4
    for (int kc = 0; kc < KCX; ++kc) {
      v16bf a  = *(const v16bf*)(xt + kc * 512 + lane * 16);
      v16bf b0 = *(const v16bf*)(wi + (0 * KCX + kc) * 512 + lane * 16);
      v16bf b1 = *(const v16bf*)(wi + (1 * KCX + kc) * 512 + lane * 16);
      v16bf b2 = *(const v16bf*)(wi + (2 * KCX + kc) * 512 + lane * 16);
      v16bf b3 = *(const v16bf*)(wi + (3 * KCX + kc) * 512 + lane * 16);
      a0 = wmma_bf16(a, b0, a0);
      a1 = wmma_bf16(a, b1, a1);
      a2 = wmma_bf16(a, b2, a2);
      a3 = wmma_bf16(a, b3, a3);
    }

    // --- LSTM cell update; C/D layout: VGPR r -> row r / r+8 ---
    __bf16* hwv = hspin + ((size_t)((bt * 2 + ((t + 1) & 1)) * 16 + g)) * 512;
#pragma unroll
    for (int r = 0; r < 8; ++r) {
      float gi = 1.f / (1.f + __expf(-(a0[r] + bI)));
      float gf = 1.f / (1.f + __expf(-(a1[r] + bF)));
      float gg = tanhf(a2[r] + bG);
      float go = 1.f / (1.f + __expf(-(a3[r] + bO)));
      float cn = gf * cc[r] + gi * gg;
      cc[r] = cn;
      float hn = go * tanhf(cn);

      int m     = r + ((lane & 16) ? 8 : 0);
      int lanep = m + lp;
      __bf16 hb = (__bf16)hn;
      hwv[lanep * 16 + ew] = hb;
      if constexpr (!FINAL) {
        hout[((size_t)(bt * TSTEPS + t) * 16 + g) * 512 + lanep * 16 + ew] = hb;
      } else {
        if (t == TSTEPS - 1)
          dout[(size_t)(bt * 16 + m) * HID + j] = hn;
      }
    }

    group_barrier(bar, 16u);   // h(t+1) published to the batch group
  }
}

// ---------------------------------------------------------------------------
extern "C" void kernel_launch(void* const* d_in, const int* in_sizes, int n_in,
                              void* d_out, int out_size, void* d_ws, size_t ws_size,
                              hipStream_t stream) {
  (void)in_sizes; (void)n_in; (void)out_size; (void)ws_size;

  const float* x    = (const float*)d_in[0];
  const float* Wih0 = (const float*)d_in[1];
  const float* Whh0 = (const float*)d_in[2];
  const float* bih0 = (const float*)d_in[3];
  const float* bhh0 = (const float*)d_in[4];
  const float* mih0 = (const float*)d_in[5];
  const float* mhh0 = (const float*)d_in[6];
  const float* Wih1 = (const float*)d_in[7];
  const float* Whh1 = (const float*)d_in[8];
  const float* bih1 = (const float*)d_in[9];
  const float* bhh1 = (const float*)d_in[10];
  const float* mih1 = (const float*)d_in[11];
  const float* mhh1 = (const float*)d_in[12];

  char* ws = (char*)d_ws;
  size_t off = 0;
  auto carve = [&](size_t bytes) -> void* {
    void* p = ws + off;
    off = (off + bytes + 255) & ~(size_t)255;
    return p;
  };

  __bf16* Wih0p = (__bf16*)carve((size_t)GDIM * DIN * 2);
  __bf16* Whh0p = (__bf16*)carve((size_t)GDIM * HID * 2);
  __bf16* Wih1p = (__bf16*)carve((size_t)GDIM * HID * 2);
  __bf16* Whh1p = (__bf16*)carve((size_t)GDIM * HID * 2);
  float*  bsum0 = (float*)carve((size_t)GDIM * 4);
  float*  bsum1 = (float*)carve((size_t)GDIM * 4);
  __bf16* xpack = (__bf16*)carve((size_t)BATCH * TSTEPS * DIN * 2);  // 16 MB
  __bf16* h0pk  = (__bf16*)carve((size_t)BATCH * TSTEPS * HID * 2);  // 32 MB

  // zero-initialized region: barriers + two h double-buffers (contiguous)
  size_t zbase = off;
  unsigned* bars  = (unsigned*)carve(4 * 256);                 // 4 groups
  __bf16*   hbufA = (__bf16*)carve((size_t)4 * 2 * 16 * 512 * 2);  // 128 KB
  __bf16*   hbufB = (__bf16*)carve((size_t)4 * 2 * 16 * 512 * 2);  // 128 KB
  int zwords = (int)((off - zbase) / 4);

  const int TPB = 256;
  zero_words<<<(zwords + TPB - 1) / TPB, TPB, 0, stream>>>(bars, zwords);

  pack_weight_bf16<<<(GDIM * DIN + TPB - 1) / TPB, TPB, 0, stream>>>(Wih0, mih0, Wih0p, DIN, GDIM);
  pack_weight_bf16<<<(GDIM * HID + TPB - 1) / TPB, TPB, 0, stream>>>(Whh0, mhh0, Whh0p, HID, GDIM);
  pack_weight_bf16<<<(GDIM * HID + TPB - 1) / TPB, TPB, 0, stream>>>(Wih1, mih1, Wih1p, HID, GDIM);
  pack_weight_bf16<<<(GDIM * HID + TPB - 1) / TPB, TPB, 0, stream>>>(Whh1, mhh1, Whh1p, HID, GDIM);
  pack_bias<<<(GDIM + TPB - 1) / TPB, TPB, 0, stream>>>(bih0, bhh0, bsum0, GDIM);
  pack_bias<<<(GDIM + TPB - 1) / TPB, TPB, 0, stream>>>(bih1, bhh1, bsum1, GDIM);
  pack_x_bf16<<<(BATCH * TSTEPS * DIN + TPB - 1) / TPB, TPB, 0, stream>>>(x, xpack);

  // Layer 0: Din=256 (KCX=8); emits h for all t (A-fragment layout).
  lstm_recur<8, false><<<64, 64, 0, stream>>>(Whh0p, Wih0p, xpack, bsum0,
                                              hbufA, bars, h0pk, nullptr);
  // Layer 1: Din=512 (KCX=16); emits final hidden state (fp32).
  lstm_recur<16, true><<<64, 64, 0, stream>>>(Whh1p, Wih1p, h0pk, bsum1,
                                              hbufB, bars, nullptr,
                                              (float*)d_out);
}
